// WisdomGraphNetwork_46248207843658
// MI455X (gfx1250) — matrix operand order repaired and checked by reading, not compiled, not run on hardware
//
#include <hip/hip_runtime.h>
#include <hip/hip_bf16.h>

// Sizes from the reference
#define NN  768
#define HH  128
#define EDD 64

typedef __attribute__((ext_vector_type(16))) __bf16 v16bf;
typedef __attribute__((ext_vector_type(8)))  float  v8f;
typedef __attribute__((ext_vector_type(4)))  float  v4f;
typedef __attribute__((ext_vector_type(2)))  float  v2f;

// ---------- prep: Wcomb = We @ W1e  (64x128), btot = be@W1e + b1 ----------
__global__ void k_wcomb(const float* __restrict__ We, const float* __restrict__ W1,
                        float* __restrict__ Wcomb) {
    int idx = blockIdx.x * blockDim.x + threadIdx.x;   // 8192
    int kk = idx >> 7, n = idx & 127;
    float s = 0.f;
    for (int h = 0; h < HH; ++h) s += We[kk * HH + h] * W1[(HH + h) * HH + n];
    Wcomb[idx] = s;
}

__global__ void k_btot(const float* __restrict__ be, const float* __restrict__ W1,
                       const float* __restrict__ b1, float* __restrict__ btot) {
    int n = threadIdx.x;  // 128
    float s = b1[n];
    for (int h = 0; h < HH; ++h) s += be[h] * W1[(HH + h) * HH + n];
    btot[n] = s;
}

// Pack Wcomb (64x128 f32) into bf16 B-fragments for wmma_f32_16x16x32_bf16.
// B element e of lane l for (ntile nt, kstep ks): K = ks*32 + (e>>3)*16 + (l>>4)*8 + (e&7),
// N = nt*16 + (l&15). Fragment stored contiguously: 16 bf16 (32B) per lane.
__global__ void k_pack(const float* __restrict__ Wcomb, __bf16* __restrict__ Bfrag) {
    int idx = blockIdx.x * blockDim.x + threadIdx.x;   // 8*2*32*16 = 8192
    int e    = idx & 15;
    int lane = (idx >> 4) & 31;
    int ks   = (idx >> 9) & 1;
    int nt   = idx >> 10;
    int k = ks * 32 + (e >> 3) * 16 + (lane >> 4) * 8 + (e & 7);
    int n = nt * 16 + (lane & 15);
    Bfrag[idx] = (__bf16)Wcomb[k * HH + n];
}

__global__ void k_cnt(const int* __restrict__ adj, float* __restrict__ cnt) {
    int i = blockIdx.x * blockDim.x + threadIdx.x;
    if (i >= NN) return;
    int c = 0;
    for (int j = 0; j < NN; ++j) c += (adj[i * NN + j] > 0) ? 1 : 0;
    cnt[i] = (float)c;
}

// ---------- e_part rows [row_base, row_base + blocks*64) of the flat
// (589824 x 64) @ (64 x 128) GEMM, bf16 output. One wave per 16-row tile.
__global__ void epart_wmma(const float* __restrict__ E, const __bf16* __restrict__ Bfrag,
                           __bf16* __restrict__ ep, size_t row_base) {
    const int lane = threadIdx.x & 31;
    const int wave = threadIdx.x >> 5;
    const int rt   = blockIdx.x * 4 + wave;
    const int m = lane & 15, hk = lane >> 4;
    const float* arow = E + (row_base + (size_t)rt * 16 + m) * EDD;
    v16bf a0, a1;
    // 16-bit A 16x32 layout: element e -> K = (e>>3)*16 + hk*8 + (e&7).
    // Each (group g, lane-half) is 8 consecutive K: two NT v4f loads.
#pragma unroll
    for (int g = 0; g < 2; ++g) {
        const float* p = arow + g * 16 + hk * 8;
        v4f f0 = __builtin_nontemporal_load((const v4f*)(p));
        v4f f1 = __builtin_nontemporal_load((const v4f*)(p + 4));
        v4f f2 = __builtin_nontemporal_load((const v4f*)(p + 32));
        v4f f3 = __builtin_nontemporal_load((const v4f*)(p + 36));
#pragma unroll
        for (int q = 0; q < 4; ++q) {
            a0[g * 8 + q]     = (__bf16)f0[q];
            a0[g * 8 + 4 + q] = (__bf16)f1[q];
            a1[g * 8 + q]     = (__bf16)f2[q];
            a1[g * 8 + 4 + q] = (__bf16)f3[q];
        }
    }
#pragma unroll
    for (int nt = 0; nt < 8; ++nt) {
        v16bf b0 = *(const v16bf*)(Bfrag + ((size_t)(nt * 2 + 0) * 32 + lane) * 16);
        v16bf b1 = *(const v16bf*)(Bfrag + ((size_t)(nt * 2 + 1) * 32 + lane) * 16);
        v8f c = {};
        c = __builtin_amdgcn_wmma_f32_16x16x32_bf16(false, a0, false, b0, (short)0, c, false, false);
        c = __builtin_amdgcn_wmma_f32_16x16x32_bf16(false, a1, false, b1, (short)0, c, false, false);
#pragma unroll
        for (int r = 0; r < 8; ++r) {
            size_t orow = (size_t)rt * 16 + r + hk * 8;   // chunk-local row
            ep[orow * HH + nt * 16 + m] = (__bf16)c[r];
        }
    }
}

// ---------- fp32 GEMM via v_wmma_f32_16x16x4_f32, K fixed at 128 ----------
// C(M x Nc) = A(M x 128) @ B(128 x Nc) [+ bias]; one wave per 16x16 tile.
__global__ void gemm_wmma_f32(const float* __restrict__ A, const float* __restrict__ B,
                              const float* __restrict__ bias, float* __restrict__ C,
                              int Nc) {
    const int lane = threadIdx.x;                      // blockDim = 32
    const int m = lane & 15, hk = lane >> 4;
    const int r0 = blockIdx.x * 16, n0 = blockIdx.y * 16;
    v8f c = {};
#pragma unroll
    for (int ks = 0; ks < HH; ks += 4) {
        const int ka = ks + hk * 2;                    // A/B elem e: K = ka + e
        v2f a = *(const v2f*)(A + (size_t)(r0 + m) * HH + ka);
        v2f b = { B[(size_t)ka * Nc + n0 + m], B[(size_t)(ka + 1) * Nc + n0 + m] };
        c = __builtin_amdgcn_wmma_f32_16x16x4_f32(false, a, false, b, (short)0, c, false, false);
    }
    const float bv = bias ? bias[n0 + m] : 0.0f;
#pragma unroll
    for (int r = 0; r < 8; ++r)
        C[(size_t)(r0 + r + hk * 8) * Nc + n0 + m] = c[r] + bv;
}

// ---------- S[i,h] = sum_j mask[i,j]*relu(pi[i,h]+e_part[i,j,h]+pj[j,h]+btot[h])
// Two i per block (one per threadIdx.y slice, mask branch wave-uniform);
// each thread covers h = 2t, 2t+1 with one packed-bf16 32-bit load.
__global__ void masked_sum(const __bf16* __restrict__ ep, int i0,
                           const float* __restrict__ pi, const float* __restrict__ pj,
                           const float* __restrict__ btot, const int* __restrict__ adj,
                           float* __restrict__ S) {
    const int i  = i0 + blockIdx.x * 2 + threadIdx.y;
    const int t  = threadIdx.x;                        // 0..63
    const int h2 = t * 2;
    const v2f bb = *(const v2f*)(btot + h2);
    const v2f pp = *(const v2f*)(pi + (size_t)i * HH + h2);
    const float base0 = pp[0] + bb[0];
    const float base1 = pp[1] + bb[1];
    const unsigned* er = (const unsigned*)(ep + (size_t)(i - i0) * NN * HH) + t;
    const int* ar = adj + i * NN;
    float acc0 = 0.f, acc1 = 0.f;
#pragma unroll 4
    for (int j = 0; j < NN; ++j) {
        if (ar[j] > 0) {                               // uniform across wave
            unsigned u = er[j * (HH / 2)];
            float e0 = __uint_as_float(u << 16);           // low bf16
            float e1 = __uint_as_float(u & 0xffff0000u);   // high bf16
            v2f pjv = *(const v2f*)(pj + (size_t)j * HH + h2);
            acc0 += fmaxf(base0 + e0 + pjv[0], 0.f);
            acc1 += fmaxf(base1 + e1 + pjv[1], 0.f);
        }
    }
    S[(size_t)i * HH + h2]     = acc0;
    S[(size_t)i * HH + h2 + 1] = acc1;
}

// messages = cnt>0 ? SW2/cnt + b2 : 0   (denom = max(cnt,1) == cnt when cnt>0)
__global__ void finalize_msgs(const float* __restrict__ SW2, const float* __restrict__ b2,
                              const float* __restrict__ cnt, float* __restrict__ msgs) {
    int i = blockIdx.x, t = threadIdx.x;
    float c = cnt[i];
    msgs[i * HH + t] = (c > 0.f) ? (SW2[i * HH + t] / c + b2[t]) : 0.f;
}

__global__ void gru_update(const float* __restrict__ gx, const float* __restrict__ gh,
                           const float* __restrict__ h_in, float* __restrict__ h_out) {
    int i = blockIdx.x, t = threadIdx.x;   // 768 x 128
    const float* gxr = gx + (size_t)i * 3 * HH;
    const float* ghr = gh + (size_t)i * 3 * HH;
    float r = 1.f / (1.f + expf(-(gxr[t] + ghr[t])));
    float z = 1.f / (1.f + expf(-(gxr[HH + t] + ghr[HH + t])));
    float n = tanhf(gxr[2 * HH + t] + r * ghr[2 * HH + t]);
    float hv = h_in[i * HH + t];
    h_out[i * HH + t] = (1.f - z) * n + z * hv;
}

extern "C" void kernel_launch(void* const* d_in, const int* in_sizes, int n_in,
                              void* d_out, int out_size, void* d_ws, size_t ws_size,
                              hipStream_t stream) {
    const float* node_features = (const float*)d_in[0];
    const float* edge_features = (const float*)d_in[1];
    const int*   adjacency     = (const int*)d_in[2];
    const float* Wn  = (const float*)d_in[3];
    const float* bn  = (const float*)d_in[4];
    const float* We  = (const float*)d_in[5];
    const float* be  = (const float*)d_in[6];
    const float* W1  = (const float*)d_in[7];
    const float* b1  = (const float*)d_in[8];
    const float* W2  = (const float*)d_in[9];
    const float* b2  = (const float*)d_in[10];
    const float* Wih = (const float*)d_in[11];
    const float* bih = (const float*)d_in[12];
    const float* Whh = (const float*)d_in[13];
    const float* bhh = (const float*)d_in[14];
    float* out = (float*)d_out;

    // Small buffers first (always fit); e_part takes the remainder of d_ws.
    char* ws = (char*)d_ws;
    size_t off = 0;
    auto take = [&](size_t b) { char* p = ws + off; off = (off + b + 255) & ~(size_t)255; return p; };
    float*  Wcomb = (float*) take(EDD * HH * 4);
    __bf16* Bfrag = (__bf16*)take(8 * 2 * 32 * 16 * 2);
    float*  btot  = (float*) take(HH * 4);
    float*  cnt   = (float*) take(NN * 4);
    float*  h     = (float*) take((size_t)NN * HH * 4);
    float*  pi    = (float*) take((size_t)NN * HH * 4);
    float*  pj    = (float*) take((size_t)NN * HH * 4);
    float*  gh    = (float*) take((size_t)NN * 3 * HH * 4);
    float*  S     = (float*) take((size_t)NN * HH * 4);
    float*  SW2   = (float*) take((size_t)NN * HH * 4);
    float*  msgs  = (float*) take((size_t)NN * HH * 4);
    float*  gx    = (float*) take((size_t)NN * 3 * HH * 4);
    __bf16* epart = (__bf16*)(ws + off);

    // Chunk size over i (multiples of 16): full 768 -> 151 MB, L2-resident,
    // computed once. Smaller ws -> recompute per chunk per step (29 GFLOP total).
    const size_t perI = (size_t)NN * HH * 2;           // e_part bytes per i
    size_t avail = (ws_size > off) ? (ws_size - off) : 0;
    int ci = (int)((avail / perI) / 16) * 16;
    if (ci > NN) ci = NN;
    if (ci < 16) ci = 16;                              // minimum ~3 MB assumed
    const bool full = (ci >= NN);

    // Fold We@W1e and be@W1e+b1; pack weights into WMMA B-fragment order.
    k_wcomb<<<64, 128, 0, stream>>>(We, W1, Wcomb);
    k_btot<<<1, 128, 0, stream>>>(be, W1, b1, btot);
    k_pack<<<64, 128, 0, stream>>>(Wcomb, Bfrag);
    k_cnt<<<6, 128, 0, stream>>>(adjacency, cnt);

    if (full)  // e_part (589824 x 64) @ (64 x 128), bf16 WMMA, once
        epart_wmma<<<NN * 12, 128, 0, stream>>>(edge_features, Bfrag, epart, 0);

    // h0 = node_features @ Wn + bn  (fp32 WMMA)
    gemm_wmma_f32<<<dim3(NN / 16, HH / 16), 32, 0, stream>>>(node_features, Wn, bn, h, HH);

    for (int step = 0; step < 3; ++step) {
        gemm_wmma_f32<<<dim3(NN / 16, HH / 16), 32, 0, stream>>>(h, W1, nullptr, pi, HH);               // W1i
        gemm_wmma_f32<<<dim3(NN / 16, HH / 16), 32, 0, stream>>>(h, W1 + 2 * HH * HH, nullptr, pj, HH); // W1j
        gemm_wmma_f32<<<dim3(NN / 16, 3 * HH / 16), 32, 0, stream>>>(h, Whh, bhh, gh, 3 * HH);
        if (full) {
            masked_sum<<<dim3(NN / 2), dim3(64, 2), 0, stream>>>(epart, 0, pi, pj, btot, adjacency, S);
        } else {
            for (int i0 = 0; i0 < NN; i0 += ci) {
                int cc = (NN - i0 < ci) ? (NN - i0) : ci;
                epart_wmma<<<cc * 12, 128, 0, stream>>>(edge_features, Bfrag, epart,
                                                        (size_t)i0 * NN);
                masked_sum<<<dim3(cc / 2), dim3(64, 2), 0, stream>>>(epart, i0, pi, pj, btot,
                                                                     adjacency, S);
            }
        }
        gemm_wmma_f32<<<dim3(NN / 16, HH / 16), 32, 0, stream>>>(S, W2, nullptr, SW2, HH);
        finalize_msgs<<<NN, HH, 0, stream>>>(SW2, b2, cnt, msgs);
        gemm_wmma_f32<<<dim3(NN / 16, 3 * HH / 16), 32, 0, stream>>>(msgs, Wih, bih, gx, 3 * HH);
        gru_update<<<NN, HH, 0, stream>>>(gx, gh, h, (step == 2) ? out : h);
    }
}
